// GraphConvDecoderBlock_59863254172148
// MI455X (gfx1250) — matrix operand order repaired and checked by reading, not compiled, not run on hardware
//
#include <hip/hip_runtime.h>
#include <hip/hip_bf16.h>
#include <math.h>

// ---------------------------------------------------------------------------
// GraphConvDecoderBlock for MI455X (gfx1250), wave32 + bf16 WMMA.
// All GEMMs use v_wmma_f32_16x16x32_bf16 (f32 accumulate).
// KNN MLP factored:  (lv-q)@W1 + q@W2  ==  (nv@W1)[idx] + nq@(W2-W1)
// GEMM: 32x64 tile per wave (2 A-frags reuse each B-frag, ~21 flops/byte).
// Attention: 32 queries per wave (2 Q-frags reuse each K/V frag) -> halves
// the K/V L2 re-stream, which dominates the op's memory budget.
// ---------------------------------------------------------------------------

typedef __bf16 bf16_t;
typedef __attribute__((ext_vector_type(16))) __bf16 v16bf;
typedef __attribute__((ext_vector_type(8)))  __bf16 v8bf;
typedef __attribute__((ext_vector_type(8)))  float  v8f;

union AFrag { v16bf v; v8bf h[2]; };

__device__ __forceinline__ bf16_t f2bf(float f) {
    union { float f; unsigned u; } a; a.f = f;
    unsigned r = a.u + 0x7FFFu + ((a.u >> 16) & 1u);   // RNE
    unsigned short s = (unsigned short)(r >> 16);
    union { unsigned short u; bf16_t b; } o; o.u = s;
    return o.b;
}

#define CDIM 384
#define NQ   1024
#define NV   512
#define NB   16
#define NH   6
#define DH   64
#define HIDD 1536
#define BN   (NB*NQ)     // 16384
#define BM   (NB*NV)     // 8192
#define PSTRIDE 40       // LDS row stride (bf16) for P transpose, bank-friendly

// ---------------------------------------------------------------------------
// Weight convert+transpose: Wt[o][i] = (bf16) W[i][o]
// ---------------------------------------------------------------------------
__global__ void convT_kernel(const float* __restrict__ W, bf16_t* __restrict__ Wt,
                             int In, int Out) {
    int i = blockIdx.x * blockDim.x + threadIdx.x;
    if (i >= In * Out) return;
    int o = i / In, ii = i % In;
    Wt[(size_t)o * In + ii] = f2bf(W[(size_t)ii * Out + o]);
}

// knn_w (2C,C):  W1t[o][i]=W[i][o],  Wdt[o][i]=W[C+i][o]-W[i][o]
__global__ void conv_knn_kernel(const float* __restrict__ W,
                                bf16_t* __restrict__ W1t, bf16_t* __restrict__ Wdt) {
    int i = blockIdx.x * blockDim.x + threadIdx.x;
    if (i >= CDIM * CDIM) return;
    int o = i / CDIM, ii = i % CDIM;
    float w1 = W[(size_t)ii * CDIM + o];
    float w2 = W[(size_t)(CDIM + ii) * CDIM + o];
    W1t[(size_t)o * CDIM + ii] = f2bf(w1);
    Wdt[(size_t)o * CDIM + ii] = f2bf(w2 - w1);
}

// ---------------------------------------------------------------------------
// LayerNorm over C=384, one 128-thread block per row, bf16 out
// ---------------------------------------------------------------------------
__global__ void __launch_bounds__(128) ln_kernel(
    const float* __restrict__ x, const float* __restrict__ g,
    const float* __restrict__ bta, bf16_t* __restrict__ out) {
    int row = blockIdx.x, t = threadIdx.x;
    const float* xr = x + (size_t)row * CDIM;
    float v[3], s = 0.f;
    #pragma unroll
    for (int j = 0; j < 3; ++j) { v[j] = xr[t + j * 128]; s += v[j]; }
    __shared__ float red[128];
    red[t] = s; __syncthreads();
    for (int o = 64; o > 0; o >>= 1) { if (t < o) red[t] += red[t + o]; __syncthreads(); }
    float mean = red[0] * (1.f / CDIM);
    __syncthreads();
    float s2 = 0.f;
    #pragma unroll
    for (int j = 0; j < 3; ++j) { float d = v[j] - mean; s2 += d * d; }
    red[t] = s2; __syncthreads();
    for (int o = 64; o > 0; o >>= 1) { if (t < o) red[t] += red[t + o]; __syncthreads(); }
    float rstd = rsqrtf(red[0] * (1.f / CDIM) + 1e-5f);
    #pragma unroll
    for (int j = 0; j < 3; ++j) {
        int c = t + j * 128;
        out[(size_t)row * CDIM + c] = f2bf((v[j] - mean) * rstd * g[c] + bta[c]);
    }
}

// ---------------------------------------------------------------------------
// Generic bf16 WMMA GEMM: out(M,N) = A(M,K) @ Bt(N,K)^T [+bias][gelu][+resid]
// One wave per 32x64 tile; each B-frag reused by 2 A-frags.
// ---------------------------------------------------------------------------
__global__ void __launch_bounds__(32) gemm_bf16_kernel(
    const bf16_t* __restrict__ A, const bf16_t* __restrict__ Bt,
    const float* __restrict__ bias, const float* __restrict__ resid,
    float* __restrict__ outF, bf16_t* __restrict__ outB,
    int M, int N, int K, int act) {
    const int l = threadIdx.x, r = l & 15, hh = l >> 4;
    const int m0 = blockIdx.y * 32;
    const int n0 = blockIdx.x * 64;
    v8f zero = {};
    v8f acc[2][4] = {{zero, zero, zero, zero}, {zero, zero, zero, zero}};
    for (int k = 0; k < K; k += 32) {
        AFrag a[2];
        #pragma unroll
        for (int mi = 0; mi < 2; ++mi) {
            const bf16_t* ar = A + (size_t)(m0 + mi * 16 + r) * K + k;
            a[mi].h[0] = *(const v8bf*)(ar + hh * 16);
            a[mi].h[1] = *(const v8bf*)(ar + hh * 16 + 8);
        }
        #pragma unroll
        for (int t = 0; t < 4; ++t) {
            const bf16_t* br = Bt + (size_t)(n0 + t * 16 + r) * K + k + hh * 16;
            v16bf bv = *(const v16bf*)br;
            acc[0][t] = __builtin_amdgcn_wmma_f32_16x16x32_bf16(
                false, a[0].v, false, bv, (short)0, acc[0][t], false, false);
            acc[1][t] = __builtin_amdgcn_wmma_f32_16x16x32_bf16(
                false, a[1].v, false, bv, (short)0, acc[1][t], false, false);
        }
    }
    #pragma unroll
    for (int mi = 0; mi < 2; ++mi)
        #pragma unroll
        for (int t = 0; t < 4; ++t) {
            int col = n0 + t * 16 + r;
            float bv = bias ? bias[col] : 0.f;
            #pragma unroll
            for (int i = 0; i < 8; ++i) {
                int row = m0 + mi * 16 + i + 8 * hh;
                float v = acc[mi][t][i] + bv;
                if (act == 1) v = 0.5f * v * (1.f + erff(v * 0.70710678118654752f));
                size_t oi = (size_t)row * N + col;
                if (resid) v += resid[oi];
                if (outF) outF[oi] = v;
                if (outB) outB[oi] = f2bf(v);
            }
        }
}

// ---------------------------------------------------------------------------
// qkv (BN,1152) f32 -> qh/kh (b,h,n,d) bf16 and vT (b,h,d,n) bf16
// ---------------------------------------------------------------------------
__global__ void qkv_reshape_kernel(const float* __restrict__ y,
                                   bf16_t* __restrict__ qh, bf16_t* __restrict__ kh,
                                   bf16_t* __restrict__ vT) {
    size_t i = (size_t)blockIdx.x * blockDim.x + threadIdx.x;
    const size_t total = (size_t)BN * (3 * CDIM);
    if (i >= total) return;
    int c = (int)(i % (3 * CDIM));
    size_t bn = i / (3 * CDIM);
    int b = (int)(bn >> 10), n = (int)(bn & 1023);
    int s = c / CDIM, rem = c % CDIM, h = rem >> 6, d = rem & 63;
    bf16_t val = f2bf(y[i]);
    size_t bh = (size_t)(b * NH + h);
    if (s == 0)      qh[(bh * NQ + n) * DH + d] = val;
    else if (s == 1) kh[(bh * NQ + n) * DH + d] = val;
    else             vT[(bh * DH + d) * NQ + n] = val;
}

// ---------------------------------------------------------------------------
// Flash attention: one wave per (b,h,32 query rows). K stored (m,d),
// V stored transposed (d,m). P transposed via LDS (D-layout -> A-layout).
// Two Q groups share every K/V fragment (halves K/V traffic).
// ---------------------------------------------------------------------------
__global__ void __launch_bounds__(32) attn_kernel(
    const bf16_t* __restrict__ qh, const bf16_t* __restrict__ kh,
    const bf16_t* __restrict__ vT, bf16_t* __restrict__ sa) {
    const int l = threadIdx.x, r = l & 15, hh = l >> 4;
    const int n0 = blockIdx.x * 32;
    const int h = blockIdx.y, b = blockIdx.z;
    const size_t bh = (size_t)b * NH + h;
    const bf16_t* qb = qh + bh * NQ * DH;
    const bf16_t* kb = kh + bh * NQ * DH;
    const bf16_t* vb = vT + bh * DH * NQ;

    __shared__ bf16_t Pt[32 * PSTRIDE];

    AFrag qa[2][2];   // [query-group][kk]
    #pragma unroll
    for (int qi = 0; qi < 2; ++qi)
        #pragma unroll
        for (int kk = 0; kk < 2; ++kk) {
            const bf16_t* qrow = qb + (size_t)(n0 + qi * 16 + r) * DH + kk * 32;
            qa[qi][kk].h[0] = *(const v8bf*)(qrow + hh * 16);
            qa[qi][kk].h[1] = *(const v8bf*)(qrow + hh * 16 + 8);
        }

    v8f zero = {};
    v8f o[2][4] = {{zero, zero, zero, zero}, {zero, zero, zero, zero}};
    float mrow[2][8], lrow[2][8];
    #pragma unroll
    for (int qi = 0; qi < 2; ++qi)
        #pragma unroll
        for (int i = 0; i < 8; ++i) { mrow[qi][i] = -1e30f; lrow[qi][i] = 0.f; }

    for (int m0 = 0; m0 < NQ; m0 += 32) {
        v8f s[2][2] = {{zero, zero}, {zero, zero}};   // [qi][m-subtile]
        #pragma unroll
        for (int kk = 0; kk < 2; ++kk) {
            v16bf b0 = *(const v16bf*)(kb + (size_t)(m0 + r) * DH + kk * 32 + hh * 16);
            v16bf b1 = *(const v16bf*)(kb + (size_t)(m0 + 16 + r) * DH + kk * 32 + hh * 16);
            #pragma unroll
            for (int qi = 0; qi < 2; ++qi) {
                s[qi][0] = __builtin_amdgcn_wmma_f32_16x16x32_bf16(
                    false, qa[qi][kk].v, false, b0, (short)0, s[qi][0], false, false);
                s[qi][1] = __builtin_amdgcn_wmma_f32_16x16x32_bf16(
                    false, qa[qi][kk].v, false, b1, (short)0, s[qi][1], false, false);
            }
        }
        #pragma unroll
        for (int qi = 0; qi < 2; ++qi) {
            s[qi][0] *= 0.125f; s[qi][1] *= 0.125f;   // D^-0.5
            #pragma unroll
            for (int i = 0; i < 8; ++i) {
                float mx = fmaxf(s[qi][0][i], s[qi][1][i]);
                #pragma unroll
                for (int off = 8; off >= 1; off >>= 1) mx = fmaxf(mx, __shfl_xor(mx, off, 32));
                float mn = fmaxf(mrow[qi][i], mx);
                float corr = __expf(mrow[qi][i] - mn);
                mrow[qi][i] = mn;
                float e0 = __expf(s[qi][0][i] - mn);
                float e1 = __expf(s[qi][1][i] - mn);
                s[qi][0][i] = e0; s[qi][1][i] = e1;
                float ls = e0 + e1;
                #pragma unroll
                for (int off = 8; off >= 1; off >>= 1) ls += __shfl_xor(ls, off, 32);
                lrow[qi][i] = lrow[qi][i] * corr + ls;
                o[qi][0][i] *= corr; o[qi][1][i] *= corr;
                o[qi][2][i] *= corr; o[qi][3][i] *= corr;
            }
            // P: D-layout regs -> LDS (padded stride = conflict-free readback)
            #pragma unroll
            for (int i = 0; i < 8; ++i) {
                Pt[(qi * 16 + i + 8 * hh) * PSTRIDE + r]      = f2bf(s[qi][0][i]);
                Pt[(qi * 16 + i + 8 * hh) * PSTRIDE + 16 + r] = f2bf(s[qi][1][i]);
            }
        }
        __syncthreads();
        AFrag pa[2];
        #pragma unroll
        for (int qi = 0; qi < 2; ++qi) {
            pa[qi].h[0] = *(const v8bf*)(&Pt[(qi * 16 + r) * PSTRIDE + hh * 16]);
            pa[qi].h[1] = *(const v8bf*)(&Pt[(qi * 16 + r) * PSTRIDE + hh * 16 + 8]);
        }
        #pragma unroll
        for (int dt = 0; dt < 4; ++dt) {
            v16bf bv = *(const v16bf*)(vb + (size_t)(dt * 16 + r) * NQ + m0 + hh * 16);
            #pragma unroll
            for (int qi = 0; qi < 2; ++qi)
                o[qi][dt] = __builtin_amdgcn_wmma_f32_16x16x32_bf16(
                    false, pa[qi].v, false, bv, (short)0, o[qi][dt], false, false);
        }
        __syncthreads();
    }

    #pragma unroll
    for (int qi = 0; qi < 2; ++qi)
        #pragma unroll
        for (int dt = 0; dt < 4; ++dt)
            #pragma unroll
            for (int i = 0; i < 8; ++i) {
                int n = n0 + qi * 16 + i + 8 * hh;
                float val = o[qi][dt][i] / lrow[qi][i];
                sa[((size_t)b * NQ + n) * CDIM + h * DH + dt * 16 + r] = f2bf(val);
            }
}

// ---------------------------------------------------------------------------
// KNN top-10 (smallest squared distance), v_pos staged in LDS
// ---------------------------------------------------------------------------
__global__ void __launch_bounds__(256) knn_kernel(
    const float* __restrict__ qp, const float* __restrict__ vp,
    int* __restrict__ idxo) {
    __shared__ float s[NV * 3];
    int b = blockIdx.x;
    for (int i = threadIdx.x; i < NV * 3; i += 256) s[i] = vp[(size_t)b * NV * 3 + i];
    __syncthreads();
    int n = blockIdx.y * 256 + threadIdx.x;
    const float* q = qp + ((size_t)b * NQ + n) * 3;
    float qx = q[0], qy = q[1], qz = q[2];
    float bd[10]; int bi[10];
    #pragma unroll
    for (int k = 0; k < 10; ++k) { bd[k] = 3.4e38f; bi[k] = 0; }
    for (int m = 0; m < NV; ++m) {
        float dx = s[m * 3] - qx, dy = s[m * 3 + 1] - qy, dz = s[m * 3 + 2] - qz;
        float d = dx * dx + dy * dy + dz * dz;
        if (d < bd[9]) {
            bd[9] = d; bi[9] = m;
            #pragma unroll
            for (int j = 9; j > 0; --j) {
                if (bd[j] < bd[j - 1]) {   // strict: ties keep earlier index first
                    float td = bd[j]; bd[j] = bd[j - 1]; bd[j - 1] = td;
                    int ti = bi[j]; bi[j] = bi[j - 1]; bi[j - 1] = ti;
                }
            }
        }
    }
    #pragma unroll
    for (int k = 0; k < 10; ++k) idxo[((size_t)b * NQ + n) * 10 + k] = bi[k];
}

// ---------------------------------------------------------------------------
// q2 = q1 + max_k leaky_relu( nvW[idx[k]] + qcomb )
// ---------------------------------------------------------------------------
__global__ void __launch_bounds__(384) gather_max_kernel(
    const float* __restrict__ nvW, const float* __restrict__ qcomb,
    const float* __restrict__ q1, const int* __restrict__ idx,
    float* __restrict__ q2) {
    int bn = blockIdx.x;
    int b = bn >> 10;
    int c = threadIdx.x;
    __shared__ int ii[10];
    if (c < 10) ii[c] = idx[(size_t)bn * 10 + c];
    __syncthreads();
    float base = qcomb[(size_t)bn * CDIM + c];
    float mx = -3.4e38f;
    #pragma unroll
    for (int k = 0; k < 10; ++k) {
        float t = nvW[((size_t)b * NV + ii[k]) * CDIM + c] + base;
        t = (t > 0.f) ? t : 0.2f * t;
        mx = fmaxf(mx, t);
    }
    q2[(size_t)bn * CDIM + c] = q1[(size_t)bn * CDIM + c] + mx;
}

// ---------------------------------------------------------------------------
extern "C" void kernel_launch(void* const* d_in, const int* in_sizes, int n_in,
                              void* d_out, int out_size, void* d_ws, size_t ws_size,
                              hipStream_t stream) {
    (void)in_sizes; (void)n_in; (void)out_size; (void)ws_size;
    const float* q      = (const float*)d_in[0];
    const float* v      = (const float*)d_in[1];
    const float* q_pos  = (const float*)d_in[2];
    const float* v_pos  = (const float*)d_in[3];
    const float* n1g    = (const float*)d_in[4];
    const float* n1b    = (const float*)d_in[5];
    const float* qkv_w  = (const float*)d_in[6];
    const float* proj_w = (const float*)d_in[7];
    const float* proj_b = (const float*)d_in[8];
    const float* nqg    = (const float*)d_in[9];
    const float* nqb    = (const float*)d_in[10];
    const float* nvg    = (const float*)d_in[11];
    const float* nvb    = (const float*)d_in[12];
    const float* knn_w  = (const float*)d_in[13];
    const float* knn_b  = (const float*)d_in[14];
    const float* n2g    = (const float*)d_in[15];
    const float* n2b    = (const float*)d_in[16];
    const float* fc1_w  = (const float*)d_in[17];
    const float* fc1_b  = (const float*)d_in[18];
    const float* fc2_w  = (const float*)d_in[19];
    const float* fc2_b  = (const float*)d_in[20];

    char* w = (char*)d_ws;
    size_t off = 0;
    auto alloc = [&](size_t bytes) -> void* {
        void* p = w + off;
        off = (off + bytes + 255) & ~(size_t)255;
        return p;
    };
    bf16_t* qkvT  = (bf16_t*)alloc((size_t)3 * CDIM * CDIM * 2);      // (1152,384)
    bf16_t* projT = (bf16_t*)alloc((size_t)CDIM * CDIM * 2);
    bf16_t* w1T   = (bf16_t*)alloc((size_t)CDIM * CDIM * 2);
    bf16_t* wdT   = (bf16_t*)alloc((size_t)CDIM * CDIM * 2);
    bf16_t* fc1T  = (bf16_t*)alloc((size_t)HIDD * CDIM * 2);          // (1536,384)
    bf16_t* fc2T  = (bf16_t*)alloc((size_t)CDIM * HIDD * 2);          // (384,1536)
    bf16_t* x_bf  = (bf16_t*)alloc((size_t)BN * CDIM * 2);
    float*  qkvF  = (float*) alloc((size_t)BN * 3 * CDIM * 4);        // 75.5 MB
    bf16_t* qh    = (bf16_t*)alloc((size_t)NB * NH * NQ * DH * 2);
    bf16_t* kh    = (bf16_t*)alloc((size_t)NB * NH * NQ * DH * 2);
    bf16_t* vT    = (bf16_t*)alloc((size_t)NB * NH * DH * NQ * 2);
    bf16_t* h1_bf = (bf16_t*)alloc((size_t)BN * HIDD * 2);
    int*    idx   = (int*)   alloc((size_t)BN * 10 * 4);

    // Aliases (lifetimes are stream-ordered and disjoint):
    float*  q1    = qkvF;                         // after reshape, qkvF is free
    float*  qcomb = qkvF + (size_t)BN * CDIM;
    float*  q2    = qkvF + (size_t)2 * BN * CDIM;
    bf16_t* sa_bf = x_bf;                         // x consumed by qkv gemm
    bf16_t* nq_bf = qh;                           // q/k/v consumed by attention
    bf16_t* nv_bf = kh;
    float*  nvW   = (float*)vT;
    bf16_t* hln_bf = x_bf;                        // sa consumed by proj gemm

    // 1. weight convert+transpose to bf16
    convT_kernel<<<(3*CDIM*CDIM + 255)/256, 256, 0, stream>>>(qkv_w, qkvT, CDIM, 3*CDIM);
    convT_kernel<<<(CDIM*CDIM + 255)/256, 256, 0, stream>>>(proj_w, projT, CDIM, CDIM);
    convT_kernel<<<(CDIM*HIDD + 255)/256, 256, 0, stream>>>(fc1_w, fc1T, CDIM, HIDD);
    convT_kernel<<<(HIDD*CDIM + 255)/256, 256, 0, stream>>>(fc2_w, fc2T, HIDD, CDIM);
    conv_knn_kernel<<<(CDIM*CDIM + 255)/256, 256, 0, stream>>>(knn_w, w1T, wdT);

    // 2. x = LN(q)
    ln_kernel<<<BN, 128, 0, stream>>>(q, n1g, n1b, x_bf);
    // 3. qkv = x @ qkv_w
    gemm_bf16_kernel<<<dim3(3*CDIM/64, BN/32), 32, 0, stream>>>(
        x_bf, qkvT, nullptr, nullptr, qkvF, nullptr, BN, 3*CDIM, CDIM, 0);
    // 4. reshape into qh / kh / vT(bhdn)
    qkv_reshape_kernel<<<(unsigned)(((size_t)BN*3*CDIM + 255)/256), 256, 0, stream>>>(
        qkvF, qh, kh, vT);
    // 5. attention -> sa (BN, C) bf16
    attn_kernel<<<dim3(NQ/32, NH, NB), 32, 0, stream>>>(qh, kh, vT, sa_bf);
    // 6. q1 = q + sa @ proj_w + proj_b
    gemm_bf16_kernel<<<dim3(CDIM/64, BN/32), 32, 0, stream>>>(
        sa_bf, projT, proj_b, q, q1, nullptr, BN, CDIM, CDIM, 0);
    // 7. nq = LN(q1), nv = LN(v)
    ln_kernel<<<BN, 128, 0, stream>>>(q1, nqg, nqb, nq_bf);
    ln_kernel<<<BM, 128, 0, stream>>>(v, nvg, nvb, nv_bf);
    // 8. nvW = nv @ W1 ;  9. qcomb = nq @ (W2-W1) + knn_b
    gemm_bf16_kernel<<<dim3(CDIM/64, BM/32), 32, 0, stream>>>(
        nv_bf, w1T, nullptr, nullptr, nvW, nullptr, BM, CDIM, CDIM, 0);
    gemm_bf16_kernel<<<dim3(CDIM/64, BN/32), 32, 0, stream>>>(
        nq_bf, wdT, knn_b, nullptr, qcomb, nullptr, BN, CDIM, CDIM, 0);
    // 10. KNN indices
    knn_kernel<<<dim3(NB, NQ/256), 256, 0, stream>>>(q_pos, v_pos, idx);
    // 11. q2 = q1 + max_k leaky(nvW[idx] + qcomb)
    gather_max_kernel<<<BN, CDIM, 0, stream>>>(nvW, qcomb, q1, idx, q2);
    // 12. hln = LN(q2)
    ln_kernel<<<BN, 128, 0, stream>>>(q2, n2g, n2b, hln_bf);
    // 13. h1 = gelu(hln @ fc1 + b1) (bf16)
    gemm_bf16_kernel<<<dim3(HIDD/64, BN/32), 32, 0, stream>>>(
        hln_bf, fc1T, fc1_b, nullptr, nullptr, h1_bf, BN, HIDD, CDIM, 1);
    // 14. out = q2 + h1 @ fc2 + b2
    gemm_bf16_kernel<<<dim3(CDIM/64, BN/32), 32, 0, stream>>>(
        h1_bf, fc2T, fc2_b, q2, (float*)d_out, nullptr, BN, CDIM, HIDD, 0);
}
